// BertDiora_54039278519126
// MI455X (gfx1250) — matrix operand order repaired and checked by reading, not compiled, not run on hardware
//
#include <hip/hip_runtime.h>
#include <math.h>

// ---------------- problem constants ----------------
#define NN 40
#define BB 32
#define DD 768
#define TT 820           // (N+1)*N/2
#define MAXPS 1560       // max pairs per level: outside L=1 -> 40*39
#define MAXROWS (MAXPS * BB)

typedef __bf16 bf16_t;
typedef __attribute__((ext_vector_type(16))) __bf16 v16bf;
typedef __attribute__((ext_vector_type(8)))  __bf16 v8bf;
typedef __attribute__((ext_vector_type(8)))  float  v8f;

// ---------------- helpers ----------------
__device__ __forceinline__ bf16_t f2bf(float f) {
  unsigned u = __float_as_uint(f);
  unsigned r = (u + 0x7FFFu + ((u >> 16) & 1u)) >> 16;
  unsigned short s = (unsigned short)r;
  bf16_t b;
  __builtin_memcpy(&b, &s, 2);
  return b;
}
__device__ __forceinline__ float bf2f(bf16_t b) {
  unsigned short s;
  __builtin_memcpy(&s, &b, 2);
  return __uint_as_float(((unsigned)s) << 16);
}
// triangular chart index: cells of span length L are contiguous.
__device__ __forceinline__ int ftri(int b, int e) {
  int L = e - b;
  return TT - (NN - L + 2) * (NN - L + 1) / 2 + b;
}
__device__ __forceinline__ void inside_idx(int p, int s, int L, int& li, int& ri) {
  li = ftri(p, p + s + 1);
  ri = ftri(p + s + 1, p + L);
}
__device__ __forceinline__ void outside_idx(int p, int j, int L, int& pc, int& sc) {
  if (j < p) { pc = ftri(j, p + L); sc = ftri(j, p); }
  else { int c = p + L + 1 + (j - p); pc = ftri(p, c); sc = ftri(p + L, c); }
}

// ---------------- elementwise / init kernels ----------------
__global__ void k_f2bf(bf16_t* __restrict__ dst, const float* __restrict__ src, long n) {
  long i = (long)blockIdx.x * 256 + threadIdx.x;
  if (i < n) dst[i] = f2bf(src[i]);
}
__global__ void k_transpose_f2bf(bf16_t* __restrict__ dst, const float* __restrict__ src, int R, int C) {
  long i = (long)blockIdx.x * 256 + threadIdx.x;
  if (i >= (long)R * C) return;
  int r = (int)(i / C), c = (int)(i % C);
  dst[(size_t)c * R + r] = f2bf(src[(size_t)r * C + c]);
}
__global__ void k_zero_f32(float* __restrict__ p, long n) {
  long i = (long)blockIdx.x * 256 + threadIdx.x;
  if (i < n) p[i] = 0.0f;
}
__global__ void k_init_root(bf16_t* __restrict__ ov, const float* __restrict__ rootb) {
  int b = blockIdx.x;  // 0..31
  size_t base = ((size_t)(TT - 1) * BB + b) * DD;
  for (int d = threadIdx.x; d < DD; d += 256) ov[base + d] = f2bf(rootb[d]);
}

// ---------------- WMMA GEMM: C(MxN,bf16) = relu(A(MxK,bf16) * Bt^T + bias) --
// Bt is N x K row-major (i.e., B transposed), so both LDS tiles are K-contiguous.
// Block tile 128x64, wave tile 32x64 (16 WMMAs per 64-deep K-tile).
// Double-buffered LDS fed by async global->LDS copies (ASYNCcnt pipeline).
// Callers guarantee A rows m0..m0+127 of every tile are readable (128-row
// slack), so the K-loop has NO guards; only the epilogue store is guarded.
__global__ __launch_bounds__(128)
void k_gemm_bt(const bf16_t* __restrict__ A, const bf16_t* __restrict__ Bt,
               bf16_t* __restrict__ C, const float* __restrict__ bias,
               int M, int Nc, int K, int relu)
{
  __shared__ __align__(16) bf16_t lA[2][128][72];  // 128 rows x 64 k (+8 pad)
  __shared__ __align__(16) bf16_t lB[2][64][72];   //  64 cols x 64 k (+8 pad)
  const int tid  = threadIdx.x;
  const int wave = tid >> 5, lane = tid & 31;
  const int half = lane >> 4, idx = lane & 15;
  const int m0 = blockIdx.y * 128, n0 = blockIdx.x * 64;

  v8f acc[2][4];
#pragma unroll
  for (int i = 0; i < 2; ++i)
#pragma unroll
    for (int n = 0; n < 4; ++n)
#pragma unroll
      for (int j = 0; j < 8; ++j) acc[i][n][j] = 0.0f;

  // async copy of one K-tile pair (A:128x64, B:64x64) into LDS buffer `buf`
  auto load_tiles = [&](int t, int buf) {
    const int k0 = t * 64;
#pragma unroll
    for (int it = 0; it < 8; ++it) {   // A: 1024 chunks of 8 bf16
      int c = tid + it * 128;
      int r = c >> 3, cc = (c & 7) * 8;
      unsigned lo = (unsigned)(size_t)&lA[buf][r][cc];
      unsigned long long ga =
          (unsigned long long)(size_t)(A + (size_t)(m0 + r) * K + k0 + cc);
      asm volatile("global_load_async_to_lds_b128 %0, %1, off"
                   :: "v"(lo), "v"(ga) : "memory");
    }
#pragma unroll
    for (int it = 0; it < 4; ++it) {   // B: 512 chunks of 8 bf16
      int c = tid + it * 128;
      int r = c >> 3, cc = (c & 7) * 8;
      unsigned lo = (unsigned)(size_t)&lB[buf][r][cc];
      unsigned long long ga =
          (unsigned long long)(size_t)(Bt + (size_t)(n0 + r) * K + k0 + cc);
      asm volatile("global_load_async_to_lds_b128 %0, %1, off"
                   :: "v"(lo), "v"(ga) : "memory");
    }
  };

  const int nK = K >> 6;
  int cur = 0;
  load_tiles(0, 0);
  asm volatile("s_wait_asynccnt 0x0" ::: "memory");
  __syncthreads();

  for (int t = 0; t < nK; ++t) {
    if (t + 1 < nK) load_tiles(t + 1, cur ^ 1);   // prefetch next tile async

    // Load ALL B fragments into distinct registers first so the WMMA bursts
    // are not serialized on per-fragment DS waits.
    v16bf bfr[2][4];
#pragma unroll
    for (int kk = 0; kk < 2; ++kk)
#pragma unroll
      for (int nt = 0; nt < 4; ++nt) {
        // B fragment (32x16): lane(half,idx): j0..7 -> k=8*half+j ; j8..15 -> k=16+8*half+j, n=idx
        const bf16_t* bp = &lB[cur][nt * 16 + idx][kk * 32];
        v8bf blo = *(const v8bf*)(bp + 8 * half);
        v8bf bhi = *(const v8bf*)(bp + 16 + 8 * half);
        bfr[kk][nt] = __builtin_shufflevector(blo, bhi,
            0,1,2,3,4,5,6,7,8,9,10,11,12,13,14,15);
      }

#pragma unroll
    for (int rt = 0; rt < 2; ++rt) {
      // A fragments (16x32): lane(half,idx): j0..7 -> k=8*half+j ; j8..15 -> k=16+8*half+j
      const bf16_t* ap = &lA[cur][wave * 32 + rt * 16 + idx][0];
      v8bf a0lo = *(const v8bf*)(ap + 8 * half);
      v8bf a0hi = *(const v8bf*)(ap + 16 + 8 * half);
      v16bf af0 = __builtin_shufflevector(a0lo, a0hi,
          0,1,2,3,4,5,6,7,8,9,10,11,12,13,14,15);
      v8bf a1lo = *(const v8bf*)(ap + 32 + 8 * half);
      v8bf a1hi = *(const v8bf*)(ap + 48 + 8 * half);
      v16bf af1 = __builtin_shufflevector(a1lo, a1hi,
          0,1,2,3,4,5,6,7,8,9,10,11,12,13,14,15);
#pragma unroll
      for (int nt = 0; nt < 4; ++nt)
        acc[rt][nt] = __builtin_amdgcn_wmma_f32_16x16x32_bf16(
            false, af0, false, bfr[0][nt], (short)0, acc[rt][nt], false, false);
#pragma unroll
      for (int nt = 0; nt < 4; ++nt)
        acc[rt][nt] = __builtin_amdgcn_wmma_f32_16x16x32_bf16(
            false, af1, false, bfr[1][nt], (short)0, acc[rt][nt], false, false);
    }

    asm volatile("s_wait_asynccnt 0x0" ::: "memory");
    __syncthreads();
    cur ^= 1;
  }

#pragma unroll
  for (int nt = 0; nt < 4; ++nt) {
    int col = n0 + nt * 16 + idx;
    float bv = (bias != nullptr && col < Nc) ? bias[col] : 0.0f;
#pragma unroll
    for (int rt = 0; rt < 2; ++rt)
#pragma unroll
      for (int r = 0; r < 8; ++r) {
        int row = m0 + wave * 32 + rt * 16 + half * 8 + r;  // C/D: M = r + 8*half
        if (row < M && col < Nc) {
          float v = acc[rt][nt][r] + bv;
          if (relu) v = fmaxf(v, 0.0f);
          C[(size_t)row * Nc + col] = f2bf(v);
        }
      }
  }
}

// ---------------- gather: Acat[row] = [Lc[li] , Rc[ri]] ----------------
__global__ __launch_bounds__(192)
void k_gather(const bf16_t* __restrict__ Lc, const bf16_t* __restrict__ Rc,
              bf16_t* __restrict__ Acat, int S, int L, int mode)
{
  int row = blockIdx.x;
  int p = row / (S * BB);
  int rem = row - p * (S * BB);
  int s = rem / BB, b = rem - s * BB;
  int li, ri;
  if (mode == 0) inside_idx(p, s, L, li, ri);
  else           outside_idx(p, s, L, li, ri);
  int t = threadIdx.x;  // 0..191 ; 96 chunks of 8 bf16 per half
  const bf16_t* src;
  bf16_t* dst;
  if (t < 96) {
    src = Lc + ((size_t)li * BB + b) * DD + t * 8;
    dst = Acat + (size_t)row * (2 * DD) + t * 8;
  } else {
    int u = t - 96;
    src = Rc + ((size_t)ri * BB + b) * DD + u * 8;
    dst = Acat + (size_t)row * (2 * DD) + DD + u * 8;
  }
  *(int4*)dst = *(const int4*)src;
}

// ---------------- bilinear score: sc = dot(XW[li], Y[ri]) + xsc[li] + ysc[ri]
__global__ __launch_bounds__(256)
void k_score(const bf16_t* __restrict__ XW, const bf16_t* __restrict__ Y,
             const float* __restrict__ xsc, const float* __restrict__ ysc,
             float* __restrict__ sc, int S, int L, int mode, long rows)
{
  long wid = (long)blockIdx.x * 8 + (threadIdx.x >> 5);
  if (wid >= rows) return;
  int lane = threadIdx.x & 31;
  int row = (int)wid;
  int p = row / (S * BB);
  int rem = row - p * (S * BB);
  int s = rem / BB, b = rem - s * BB;
  int li, ri;
  if (mode == 0) inside_idx(p, s, L, li, ri);
  else           outside_idx(p, s, L, li, ri);

  const bf16_t* xp = XW + ((size_t)li * BB + b) * DD;
  const bf16_t* yp = Y  + ((size_t)ri * BB + b) * DD;
  float acc = 0.0f;
  for (int c = lane; c < 96; c += 32) {
    v8bf xv = *(const v8bf*)(xp + c * 8);
    v8bf yv = *(const v8bf*)(yp + c * 8);
#pragma unroll
    for (int j = 0; j < 8; ++j) acc += bf2f(xv[j]) * bf2f(yv[j]);
  }
#pragma unroll
  for (int off = 16; off > 0; off >>= 1) acc += __shfl_xor(acc, off, 32);
  if (lane == 0)
    sc[row] = acc + xsc[(size_t)li * BB + b] + ysc[(size_t)ri * BB + b];
}

// ---------------- softmax over splits + weighted sums into chart ----------
__global__ __launch_bounds__(256)
void k_reduce(const float* __restrict__ sc, const bf16_t* __restrict__ v,
              float* __restrict__ outSc, bf16_t* __restrict__ outV,
              int S, int cell0)
{
  __shared__ float ssc[40];
  __shared__ float ssw[40];
  int p = blockIdx.x, b = blockIdx.y;
  int t = threadIdx.x;
  if (t < S) ssc[t] = sc[((size_t)p * S + t) * BB + b];
  __syncthreads();
  if (t == 0) {
    float m = -3.4e38f;
    for (int s = 0; s < S; ++s) m = fmaxf(m, ssc[s]);
    float sum = 0.0f;
    for (int s = 0; s < S; ++s) { float e = __expf(ssc[s] - m); ssw[s] = e; sum += e; }
    float inv = 1.0f / sum, ws = 0.0f;
    for (int s = 0; s < S; ++s) { ssw[s] *= inv; ws += ssw[s] * ssc[s]; }
    outSc[(size_t)(cell0 + p) * BB + b] = ws;   // relu(softmax) == softmax (>=0)
  }
  __syncthreads();
  for (int d = t; d < DD; d += 256) {
    float a = 0.0f;
    for (int s = 0; s < S; ++s)
      a += ssw[s] * bf2f(v[(((size_t)p * S + s) * BB + b) * DD + d]);
    outV[((size_t)(cell0 + p) * BB + b) * DD + d] = f2bf(a);
  }
}

// ---------------- cosine loss ----------------
__global__ __launch_bounds__(256)
void k_loss_partial(const bf16_t* __restrict__ ov, const float* __restrict__ base,
                    float* __restrict__ partial)
{
  int i = blockIdx.x * 8 + (threadIdx.x >> 5);  // 0..1279 = n*32+b (cells 0..39 are span-1)
  if (i >= NN * BB) return;
  int lane = threadIdx.x & 31;
  float num = 0.0f, tt = 0.0f, bb = 0.0f;
  const bf16_t* tp = ov + (size_t)i * DD;
  const float*  bp = base + (size_t)i * DD;
  for (int d = lane; d < DD; d += 32) {
    float tv = bf2f(tp[d]);
    float bv = bp[d];
    num += tv * bv; tt += tv * tv; bb += bv * bv;
  }
#pragma unroll
  for (int off = 16; off > 0; off >>= 1) {
    num += __shfl_xor(num, off, 32);
    tt  += __shfl_xor(tt,  off, 32);
    bb  += __shfl_xor(bb,  off, 32);
  }
  if (lane == 0) {
    float den = fmaxf(sqrtf(tt) * sqrtf(bb), 1e-8f);
    partial[i] = 1.0f - num / den;
  }
}
__global__ __launch_bounds__(256)
void k_loss_final(const float* __restrict__ partial, float* __restrict__ out)
{
  __shared__ float red[256];
  float a = 0.0f;
  for (int i = threadIdx.x; i < NN * BB; i += 256) a += partial[i];
  red[threadIdx.x] = a;
  __syncthreads();
  for (int s = 128; s > 0; s >>= 1) {
    if (threadIdx.x < s) red[threadIdx.x] += red[threadIdx.x + s];
    __syncthreads();
  }
  if (threadIdx.x == 0) out[0] = red[0] / (float)(NN * BB);
}

// ---------------- host driver ----------------
static inline size_t alignup(size_t x) { return (x + 255) & ~(size_t)255; }
static inline int ftri_h(int b, int e) {
  int L = e - b;
  return TT - (NN - L + 2) * (NN - L + 1) / 2 + b;
}

extern "C" void kernel_launch(void* const* d_in, const int* in_sizes, int n_in,
                              void* d_out, int out_size, void* d_ws, size_t ws_size,
                              hipStream_t stream) {
  (void)in_sizes; (void)n_in; (void)out_size; (void)ws_size;
  const float* base  = (const float*)d_in[0];   // N,B,D
  const float* Wbil  = (const float*)d_in[1];   // D,D
  const float* W1    = (const float*)d_in[2];   // D,2D
  const float* b1    = (const float*)d_in[3];   // D
  const float* W2    = (const float*)d_in[4];   // D,D
  const float* b2    = (const float*)d_in[5];   // D
  const float* rootb = (const float*)d_in[6];   // D

  // workspace carve-up (chart buffers get 128 rows of slack: GEMM A-tiles are
  // read unguarded in full 128-row tiles)
  char* W = (char*)d_ws;
  size_t off = 0;
  auto take = [&](size_t bytes) -> char* { char* q = W + off; off = alignup(off + bytes); return q; };
  const size_t chartE = (size_t)TT * BB * DD;
  const size_t slackE = (size_t)128 * DD;
  bf16_t* iv   = (bf16_t*)take((chartE + slackE) * 2);
  bf16_t* ivW  = (bf16_t*)take((chartE + slackE) * 2);
  bf16_t* ov   = (bf16_t*)take((chartE + slackE) * 2);
  bf16_t* ovW  = (bf16_t*)take((chartE + slackE) * 2);
  float*  isc  = (float*)take((size_t)TT * BB * 4);
  float*  osc  = (float*)take((size_t)TT * BB * 4);
  bf16_t* W1b  = (bf16_t*)take((size_t)DD * 2 * DD * 2);
  bf16_t* W2b  = (bf16_t*)take((size_t)DD * DD * 2);
  bf16_t* WbT  = (bf16_t*)take((size_t)DD * DD * 2);
  bf16_t* Acat = (bf16_t*)take((size_t)MAXROWS * 2 * DD * 2);
  bf16_t* hbuf = (bf16_t*)take((size_t)MAXROWS * DD * 2);
  bf16_t* vbuf = (bf16_t*)take((size_t)MAXROWS * DD * 2);
  float*  scb  = (float*)take((size_t)MAXROWS * 4);
  float*  lossp= (float*)take((size_t)NN * BB * 4);

  // ---- init ----
  {
    long n = (long)DD * 2 * DD;
    k_f2bf<<<dim3((unsigned)((n + 255) / 256)), 256, 0, stream>>>(W1b, W1, n);
    n = (long)DD * DD;
    k_f2bf<<<dim3((unsigned)((n + 255) / 256)), 256, 0, stream>>>(W2b, W2, n);
    k_transpose_f2bf<<<dim3((unsigned)((n + 255) / 256)), 256, 0, stream>>>(WbT, Wbil, DD, DD);
    long nsc = (long)TT * BB;
    k_zero_f32<<<dim3((unsigned)((nsc + 255) / 256)), 256, 0, stream>>>(isc, nsc);
    k_zero_f32<<<dim3((unsigned)((nsc + 255) / 256)), 256, 0, stream>>>(osc, nsc);
    long nb = (long)NN * BB * DD;
    k_f2bf<<<dim3((unsigned)((nb + 255) / 256)), 256, 0, stream>>>(iv, base, nb);  // span-1 cells = 0..N-1
    k_init_root<<<dim3(BB), 256, 0, stream>>>(ov, rootb);
    // ivW for base cells, ovW for root
    k_gemm_bt<<<dim3(DD / 64, (NN * BB + 127) / 128), 128, 0, stream>>>(
        iv, WbT, ivW, nullptr, NN * BB, DD, DD, 0);
    k_gemm_bt<<<dim3(DD / 64, 1), 128, 0, stream>>>(
        ov + (size_t)(TT - 1) * BB * DD, WbT, ovW + (size_t)(TT - 1) * BB * DD,
        nullptr, BB, DD, DD, 0);
  }

  // ---- inside pass ----
  for (int L = 2; L <= NN; ++L) {
    int P = NN + 1 - L, S = L - 1;
    long rows = (long)P * S * BB;
    int cell0 = ftri_h(0, L);
    k_gather<<<dim3((unsigned)rows), 192, 0, stream>>>(iv, iv, Acat, S, L, 0);
    dim3 g1(DD / 64, (unsigned)((rows + 127) / 128));
    k_gemm_bt<<<g1, 128, 0, stream>>>(Acat, W1b, hbuf, b1, (int)rows, DD, 2 * DD, 1);
    k_gemm_bt<<<g1, 128, 0, stream>>>(hbuf, W2b, vbuf, b2, (int)rows, DD, DD, 1);
    k_score<<<dim3((unsigned)((rows + 7) / 8)), 256, 0, stream>>>(
        ivW, iv, isc, isc, scb, S, L, 0, rows);
    k_reduce<<<dim3(P, BB), 256, 0, stream>>>(scb, vbuf, isc, iv, S, cell0);
    k_gemm_bt<<<dim3(DD / 64, (P * BB + 127) / 128), 128, 0, stream>>>(
        iv + (size_t)cell0 * BB * DD, WbT, ivW + (size_t)cell0 * BB * DD,
        nullptr, P * BB, DD, DD, 0);
  }

  // ---- outside pass ----
  for (int L = NN - 1; L >= 1; --L) {
    int P = NN + 1 - L, S = NN - L;
    long rows = (long)P * S * BB;
    int cell0 = ftri_h(0, L);
    k_gather<<<dim3((unsigned)rows), 192, 0, stream>>>(ov, iv, Acat, S, L, 1);
    dim3 g1(DD / 64, (unsigned)((rows + 127) / 128));
    k_gemm_bt<<<g1, 128, 0, stream>>>(Acat, W1b, hbuf, b1, (int)rows, DD, 2 * DD, 1);
    k_gemm_bt<<<g1, 128, 0, stream>>>(hbuf, W2b, vbuf, b2, (int)rows, DD, DD, 1);
    k_score<<<dim3((unsigned)((rows + 7) / 8)), 256, 0, stream>>>(
        ovW, iv, osc, isc, scb, S, L, 1, rows);
    k_reduce<<<dim3(P, BB), 256, 0, stream>>>(scb, vbuf, osc, ov, S, cell0);
    k_gemm_bt<<<dim3(DD / 64, (P * BB + 127) / 128), 128, 0, stream>>>(
        ov + (size_t)cell0 * BB * DD, WbT, ovW + (size_t)cell0 * BB * DD,
        nullptr, P * BB, DD, DD, 0);
  }

  // ---- loss ----
  k_loss_partial<<<dim3((NN * BB + 7) / 8), 256, 0, stream>>>(ov, base, lossp);
  k_loss_final<<<dim3(1), 256, 0, stream>>>(lossp, (float*)d_out);
}
